// m_block_1w1a_3719441678863
// MI455X (gfx1250) — 8-bit matrix operand order repaired and checked by reading, compiled, not run on hardware
//
#include <hip/hip_runtime.h>
#include <hip/hip_bf16.h>

typedef __attribute__((ext_vector_type(8)))  int   v8i;
typedef __attribute__((ext_vector_type(16))) int   v16i;
typedef __attribute__((ext_vector_type(8)))  float v8f;

#define EPSBN 1e-5f

__device__ __forceinline__ unsigned int fp8_sign(float f) {
    return (f > 0.f) ? 0x38u : ((f < 0.f) ? 0xB8u : 0x00u);
}

__device__ __forceinline__ v8f wmma_fp8_k64(v8i a, v8i b, v8f c) {
    return __builtin_amdgcn_wmma_f32_16x16x64_fp8_fp8(a, b, (short)0, c, false, false);
}

__device__ __forceinline__ v8f wmma_fp8_k128(v16i a, v16i b, v8f c) {
    return __builtin_amdgcn_wmma_f32_16x16x128_fp8_fp8(a, b, (short)0, c, false, false);
}

__device__ __forceinline__ v16i cat16(const v8i& lo, const v8i& hi) {
    v16i r;
    #pragma unroll
    for (int i = 0; i < 8; ++i) { r[i] = lo[i]; r[8 + i] = hi[i]; }
    return r;
}

__device__ __forceinline__ v8i load_B(const unsigned char* bw, int rowBytes,
                                      int n, int kofs, int lanehi) {
    const unsigned char* p = bw + n * rowBytes + kofs + lanehi * 8;
    uint2 q0 = *(const uint2*)(p);
    uint2 q1 = *(const uint2*)(p + 16);
    uint2 q2 = *(const uint2*)(p + 32);
    uint2 q3 = *(const uint2*)(p + 48);
    v8i r;
    r[0] = (int)q0.x; r[1] = (int)q0.y; r[2] = (int)q1.x; r[3] = (int)q1.y;
    r[4] = (int)q2.x; r[5] = (int)q2.y; r[6] = (int)q3.x; r[7] = (int)q3.y;
    return r;
}

__device__ __forceinline__ v8i load_A2(const unsigned char* rA, const unsigned char* rB,
                                       int m, int koff) {
    v8i a;
    uint2 q;
    const unsigned char* pa = rA + m * 32 + koff;
    q = *(const uint2*)(pa);       a[0] = (int)q.x; a[1] = (int)q.y;
    q = *(const uint2*)(pa + 16);  a[2] = (int)q.x; a[3] = (int)q.y;
    const unsigned char* pb = rB + m * 32 + koff;
    q = *(const uint2*)(pb);       a[4] = (int)q.x; a[5] = (int)q.y;
    q = *(const uint2*)(pb + 16);  a[6] = (int)q.x; a[7] = (int)q.y;
    return a;
}

__device__ __forceinline__ v8i load_A64(const unsigned char* st, int m, int koff) {
    v8i a;
    uint2 q;
    const unsigned char* p = st + m * 64 + koff;
    q = *(const uint2*)(p);       a[0] = (int)q.x; a[1] = (int)q.y;
    q = *(const uint2*)(p + 16);  a[2] = (int)q.x; a[3] = (int)q.y;
    q = *(const uint2*)(p + 32);  a[4] = (int)q.x; a[5] = (int)q.y;
    q = *(const uint2*)(p + 48);  a[6] = (int)q.x; a[7] = (int)q.y;
    return a;
}

__device__ __forceinline__ void bn_relu_store(const v8f& c, float inv, float beta, float* p) {
    float4 f0, f1;
    f0.x = fmaxf(fmaf(c[0], inv, beta), 0.f);
    f0.y = fmaxf(fmaf(c[1], inv, beta), 0.f);
    f0.z = fmaxf(fmaf(c[2], inv, beta), 0.f);
    f0.w = fmaxf(fmaf(c[3], inv, beta), 0.f);
    f1.x = fmaxf(fmaf(c[4], inv, beta), 0.f);
    f1.y = fmaxf(fmaf(c[5], inv, beta), 0.f);
    f1.z = fmaxf(fmaf(c[6], inv, beta), 0.f);
    f1.w = fmaxf(fmaf(c[7], inv, beta), 0.f);
    *(float4*)(p)     = f0;
    *(float4*)(p + 4) = f1;
}

__global__ __launch_bounds__(128) void bconv1_kernel(
    const float* __restrict__ x,  const float* __restrict__ w1,
    const float* __restrict__ g1, const float* __restrict__ b1,
    const float* __restrict__ m1, const float* __restrict__ v1,
    unsigned char* __restrict__ hpack)
{
    __shared__ __align__(16) unsigned char sBw[32 * 64];
    __shared__ float sInv[32], sBeta[32];
    __shared__ __align__(16) unsigned char sA[4][16 * 64];

    const int tid = threadIdx.x;

    for (int idx = tid; idx < 32 * 64; idx += 128)
        sBw[idx] = (unsigned char)fp8_sign(w1[idx]);
    if (tid < 32) {
        float inv = g1[tid] * rsqrtf(v1[tid] + EPSBN);
        sInv[tid]  = inv;
        sBeta[tid] = b1[tid] - m1[tid] * inv;
    }

    const int wave = tid >> 5, lane = tid & 31;
    const int tile = blockIdx.x * 4 + wave;
    const int b    = tile / 196;
    const int hw0  = (tile % 196) * 16;

    const int m = lane & 15, chalf = lane >> 4;
    unsigned char* aw = sA[wave];
    const float* xb = x + ((size_t)b * 64 + chalf * 32) * 3136 + hw0 + m;
    #pragma unroll
    for (int g = 0; g < 8; ++g) {
        unsigned int pk = 0;
        #pragma unroll
        for (int k = 0; k < 4; ++k) {
            float f = xb[(size_t)(g * 4 + k) * 3136];
            pk |= fp8_sign(f) << (8 * k);
        }
        *(unsigned int*)(aw + m * 64 + chalf * 32 + g * 4) = pk;
    }
    __syncthreads();

    const int lanehi = chalf;
    const int koff   = lanehi * 8;
    const int n      = lane & 15;
    v8i a = load_A64(aw, m, koff);

    v8i bf0 = load_B(sBw, 64, 0 * 16 + n, 0, lanehi);
    v8i bf1 = load_B(sBw, 64, 1 * 16 + n, 0, lanehi);
    v8f c0 = {0.f, 0.f, 0.f, 0.f, 0.f, 0.f, 0.f, 0.f};
    v8f c1 = {0.f, 0.f, 0.f, 0.f, 0.f, 0.f, 0.f, 0.f};
    c0 = wmma_fp8_k64(a, bf0, c0);
    c1 = wmma_fp8_k64(a, bf1, c1);

    #pragma unroll
    for (int nt = 0; nt < 2; ++nt) {
        const v8f& c = (nt == 0) ? c0 : c1;
        const int ch = nt * 16 + n;
        const float inv = sInv[ch], beta = sBeta[ch];
        unsigned char* hp = hpack + ((size_t)b * 3136 + hw0) * 32 + ch;
        #pragma unroll
        for (int r = 0; r < 8; ++r) {
            int mm = r + 8 * lanehi;
            float y = fmaf(c[r], inv, beta);
            hp[(size_t)mm * 32] = (y > 0.f) ? (unsigned char)0x38 : (unsigned char)0x00;
        }
    }
}

#define R_C 0
#define R_U 1
#define R_D 2
#define R_L 3
#define R_R 4
#define R_Z 5

__global__ __launch_bounds__(128) void bconv_rest_kernel(
    const unsigned char* __restrict__ hpack,
    const float* __restrict__ w2, const float* __restrict__ g2, const float* __restrict__ b2,
    const float* __restrict__ m2, const float* __restrict__ v2,
    const float* __restrict__ w3, const float* __restrict__ g3, const float* __restrict__ b3,
    const float* __restrict__ m3, const float* __restrict__ v3,
    const float* __restrict__ w4, const float* __restrict__ g4, const float* __restrict__ b4,
    const float* __restrict__ m4, const float* __restrict__ v4,
    float* __restrict__ out)
{
    __shared__ __align__(16) unsigned char sBw2[96 * 128];
    __shared__ __align__(16) unsigned char sBw3[96 * 128];
    __shared__ __align__(16) unsigned char sBw4[64 * 64];
    __shared__ float sInv[256], sBeta[256];
    __shared__ __align__(16) unsigned char sReg[4][6][16 * 32];

    const int tid = threadIdx.x;

    for (int idx = tid; idx < 96 * 128; idx += 128) {
        int o = idx >> 7, k = idx & 127;
        int t = k >> 5, ci = k & 31;
        sBw2[idx] = (t < 3) ? (unsigned char)fp8_sign(w2[o * 96 + ci * 3 + t]) : 0;
        sBw3[idx] = (t < 3) ? (unsigned char)fp8_sign(w3[o * 96 + ci * 3 + t]) : 0;
    }
    for (int idx = tid; idx < 64 * 64; idx += 128) {
        int o = idx >> 6, k = idx & 63;
        sBw4[idx] = (k < 32) ? (unsigned char)fp8_sign(w4[o * 32 + k]) : 0;
    }
    for (int ch = tid; ch < 256; ch += 128) {
        float g, bb, mm, vv;
        if (ch < 96)       { g = g2[ch];       bb = b2[ch];       mm = m2[ch];       vv = v2[ch]; }
        else if (ch < 192) { int c = ch - 96;  g = g3[c]; bb = b3[c]; mm = m3[c]; vv = v3[c]; }
        else               { int c = ch - 192; g = g4[c]; bb = b4[c]; mm = m4[c]; vv = v4[c]; }
        float inv = g * rsqrtf(vv + EPSBN);
        sInv[ch] = inv;
        sBeta[ch] = bb - mm * inv;
    }

    const int wave = tid >> 5, lane = tid & 31;
    const int tile = blockIdx.x * 4 + wave;
    const int b    = tile / 196;
    const int hw0  = (tile % 196) * 16;

    *(int4*)(&sReg[wave][R_Z][lane * 16]) = make_int4(0, 0, 0, 0);

    {
        const int sm = lane >> 1, half = lane & 1;
        const int p = hw0 + sm;
        const int w = p % 56;
        const unsigned char* hb = hpack + (size_t)b * 3136 * 32;
        const int4 z = make_int4(0, 0, 0, 0);
        int4 val;
        val = *(const int4*)(hb + (size_t)p * 32 + half * 16);
        *(int4*)(&sReg[wave][R_C][sm * 32 + half * 16]) = val;
        val = (p >= 56)       ? *(const int4*)(hb + (size_t)(p - 56) * 32 + half * 16) : z;
        *(int4*)(&sReg[wave][R_U][sm * 32 + half * 16]) = val;
        val = (p + 56 < 3136) ? *(const int4*)(hb + (size_t)(p + 56) * 32 + half * 16) : z;
        *(int4*)(&sReg[wave][R_D][sm * 32 + half * 16]) = val;
        val = (w != 0)        ? *(const int4*)(hb + (size_t)(p - 1) * 32 + half * 16) : z;
        *(int4*)(&sReg[wave][R_L][sm * 32 + half * 16]) = val;
        val = (w != 55)       ? *(const int4*)(hb + (size_t)(p + 1) * 32 + half * 16) : z;
        *(int4*)(&sReg[wave][R_R][sm * 32 + half * 16]) = val;
    }
    __syncthreads();

    const int m = lane & 15, lanehi = lane >> 4;
    const int koff = lanehi * 8;
    const int n = lane & 15;

    const unsigned char* rC = sReg[wave][R_C];
    const unsigned char* rU = sReg[wave][R_U];
    const unsigned char* rD = sReg[wave][R_D];
    const unsigned char* rL = sReg[wave][R_L];
    const unsigned char* rR = sReg[wave][R_R];
    const unsigned char* rZ = sReg[wave][R_Z];

    v16i a2 = cat16(load_A2(rU, rC, m, koff), load_A2(rD, rZ, m, koff));
    v16i a3 = cat16(load_A2(rL, rC, m, koff), load_A2(rR, rZ, m, koff));
    v8i  a4 = load_A2(rC, rZ, m, koff);

    float* ob = out + (size_t)b * 256 * 3136 + hw0 + lanehi * 8;

    #pragma unroll
    for (int nt = 0; nt < 6; ++nt) {
        v16i b2 = cat16(load_B(sBw2, 128, nt * 16 + n, 0,  lanehi),
                        load_B(sBw2, 128, nt * 16 + n, 64, lanehi));
        v16i b3 = cat16(load_B(sBw3, 128, nt * 16 + n, 0,  lanehi),
                        load_B(sBw3, 128, nt * 16 + n, 64, lanehi));
        v8f c2 = {0.f, 0.f, 0.f, 0.f, 0.f, 0.f, 0.f, 0.f};
        v8f c3 = {0.f, 0.f, 0.f, 0.f, 0.f, 0.f, 0.f, 0.f};
        c2 = wmma_fp8_k128(a2, b2, c2);
        c3 = wmma_fp8_k128(a3, b3, c3);
        const int ch2 = nt * 16 + n;
        const int ch3 = 96 + nt * 16 + n;
        bn_relu_store(c2, sInv[ch2], sBeta[ch2], ob + (size_t)ch2 * 3136);
        bn_relu_store(c3, sInv[ch3], sBeta[ch3], ob + (size_t)ch3 * 3136);
    }
    #pragma unroll
    for (int nt = 0; nt < 4; nt += 2) {
        v8i bf0 = load_B(sBw4, 64, (nt + 0) * 16 + n, 0, lanehi);
        v8i bf1 = load_B(sBw4, 64, (nt + 1) * 16 + n, 0, lanehi);
        v8f c0 = {0.f, 0.f, 0.f, 0.f, 0.f, 0.f, 0.f, 0.f};
        v8f c1 = {0.f, 0.f, 0.f, 0.f, 0.f, 0.f, 0.f, 0.f};
        c0 = wmma_fp8_k64(a4, bf0, c0);
        c1 = wmma_fp8_k64(a4, bf1, c1);
        const int ch0 = 192 + (nt + 0) * 16 + n;
        const int ch1 = 192 + (nt + 1) * 16 + n;
        bn_relu_store(c0, sInv[ch0], sBeta[ch0], ob + (size_t)ch0 * 3136);
        bn_relu_store(c1, sInv[ch1], sBeta[ch1], ob + (size_t)ch1 * 3136);
    }
}

extern "C" void kernel_launch(void* const* d_in, const int* in_sizes, int n_in,
                              void* d_out, int out_size, void* d_ws, size_t ws_size,
                              hipStream_t stream) {
    (void)in_sizes; (void)n_in; (void)out_size; (void)ws_size;
    const float* x  = (const float*)d_in[0];
    const float* w1 = (const float*)d_in[1];
    const float* g1 = (const float*)d_in[2];
    const float* b1 = (const float*)d_in[3];
    const float* m1 = (const float*)d_in[4];
    const float* v1 = (const float*)d_in[5];
    const float* w2 = (const float*)d_in[6];
    const float* g2 = (const float*)d_in[7];
    const float* b2 = (const float*)d_in[8];
    const float* m2 = (const float*)d_in[9];
    const float* v2 = (const float*)d_in[10];
    const float* w3 = (const float*)d_in[11];
    const float* g3 = (const float*)d_in[12];
    const float* b3 = (const float*)d_in[13];
    const float* m3 = (const float*)d_in[14];
    const float* v3 = (const float*)d_in[15];
    const float* w4 = (const float*)d_in[16];
    const float* g4 = (const float*)d_in[17];
    const float* b4 = (const float*)d_in[18];
    const float* m4 = (const float*)d_in[19];
    const float* v4 = (const float*)d_in[20];

    unsigned char* hpack = (unsigned char*)d_ws;
    float* out = (float*)d_out;

    bconv1_kernel<<<3136, 128, 0, stream>>>(x, w1, g1, b1, m1, v1, hpack);
    bconv_rest_kernel<<<3136, 128, 0, stream>>>(hpack,
                                                w2, g2, b2, m2, v2,
                                                w3, g3, b3, m3, v3,
                                                w4, g4, b4, m4, v4,
                                                out);
}
